// TemporalAttention_89395449299700
// MI455X (gfx1250) — matrix operand order repaired and checked by reading, not compiled, not run on hardware
//
#include <hip/hip_runtime.h>
#include <hip/hip_bf16.h>
#include <stdint.h>

#define DIM     512
#define NHEADS  8
#define HDIM    64
#define NF      16
#define NTOK    65536      // 4 * 32 * 32 * 16
#define NSITES  4096       // 4 * 32 * 32
#define CSTRIDE 16384      // 16*32*32 : stride between channels in x

#define KCHUNK  128        // K elements per TDM chunk
#define BROWS   64         // B-panel rows (output columns) per block
#define BSTRIDE 136        // LDS row stride in bf16: 128 + 8 pad (16B TDM pad)

typedef __attribute__((ext_vector_type(16))) __bf16 v16bf;
typedef __attribute__((ext_vector_type(8)))  float  v8f;
typedef __attribute__((ext_vector_type(4)))  unsigned int u32x4;
typedef __attribute__((ext_vector_type(8)))  int i32x8;
typedef __attribute__((ext_vector_type(4)))  int i32x4;

union FragU { v16bf v; float4 q[2]; };

// ---------------------------------------------------------------------------
// A-fragment, bf16 16x32 (M x K), row-major source with stride lda (elements).
// Layout: lanes 0-15 row M=lane, K k0..k0+15;
//         lanes 16-31 row M=lane-16, K k0+16..k0+31.
// ---------------------------------------------------------------------------
__device__ inline v16bf load_a_frag(const __bf16* A, int lda, int row0, int k0) {
  int ln = threadIdx.x & 31;
  int r  = row0 + (ln & 15);
  int kk = k0 + ((ln >> 4) << 4);           // +16 for upper half-wave
  const __bf16* p = A + (long)r * lda + kk;
  FragU fr;
  fr.q[0] = *(const float4*)p;              // K kk .. kk+7
  fr.q[1] = *(const float4*)(p + 8);        // K kk+8 .. kk+15
  return fr.v;
}

// ---------------------------------------------------------------------------
// B-fragment, bf16 32x16 (K x N), source row-major [N x K] stride ldw:
// lanes 0-15 hold K=k0..k0+15 of column N=lane, lanes 16-31 K=k0+16..k0+31.
// ---------------------------------------------------------------------------
__device__ inline v16bf load_b_frag(const __bf16* Wm, int ldw, int n0, int k0) {
  int ln = threadIdx.x & 31;
  int n  = n0 + (ln & 15);
  int kk = k0 + ((ln >> 4) << 4);
  const __bf16* p = Wm + (long)n * ldw + kk;
  FragU fr;
  fr.q[0] = *(const float4*)p;
  fr.q[1] = *(const float4*)(p + 8);
  return fr.v;
}

// B-fragment from the TDM-staged LDS panel (row stride BSTRIDE elements).
__device__ inline v16bf load_b_lds(const __bf16* bl, int n0, int k0) {
  int ln = threadIdx.x & 31;
  int n  = n0 + (ln & 15);
  int kk = k0 + ((ln >> 4) << 4);
  const __bf16* p = bl + n * BSTRIDE + kk;
  FragU fr;
  fr.q[0] = *(const float4*)p;
  fr.q[1] = *(const float4*)(p + 8);
  return fr.v;
}

__device__ inline v8f wmma_bf16(v16bf a, v16bf b, v8f c) {
  return __builtin_amdgcn_wmma_f32_16x16x32_bf16(false, a, false, b, (short)0, c,
                                                 false, false);
}
// Variant with the reuse-A OPSEL hint: legal when this WMMA is identical in
// form to the immediately preceding one and shares its A operand (ISA §7.12).
__device__ inline v8f wmma_bf16_ra(v16bf a, v16bf b, v8f c) {
  return __builtin_amdgcn_wmma_f32_16x16x32_bf16(false, a, false, b, (short)0, c,
                                                 true, false);
}

// ---------------------------------------------------------------------------
// Tensor Data Mover: 2D tile (rows x KCHUNK bf16) global -> LDS.
// D# per CDNA5 ISA:
//   group0: count=1 | lds_addr | global_addr(57b) | type=2
//   group1: data_size=1(2B), pad_enable, pad_interval=5 (256B = one row),
//           pad_amount=3 (16B), tensor dims/strides, tile_dim0=KCHUNK,
//           tile_dim1=rows
// groups 2/3 zero (2D tensor). 6-arg builtin form (clang-23 / therock-10.0).
// ---------------------------------------------------------------------------
__device__ inline void tdm_load_panel(const __bf16* gsrc, uint32_t lds_off,
                                      int rows, int tensor_rows) {
  uint64_t ga = (uint64_t)(uintptr_t)gsrc;
  u32x4 g0;
  g0[0] = 1u;                                       // count=1, user mode
  g0[1] = lds_off;                                  // LDS byte address
  g0[2] = (uint32_t)ga;                             // global_addr[31:0]
  g0[3] = (uint32_t)((ga >> 32) & 0x01FFFFFFu)      // global_addr[56:32]
        | (2u << 30);                               // type=2 (image)
  i32x8 g1;
  g1[0] = (int)((1u << 16)                          // data_size = 2 bytes
              | (1u << 20)                          // pad_enable
              | (5u << 22)                          // pad_interval: 64 DW=256B
              | (3u << 25));                        // pad_amount: 4 DW = 16B
  g1[1] = (int)(((uint32_t)DIM & 0xFFFFu) << 16);   // tensor_dim0[15:0] = 512
  g1[2] = (int)(((uint32_t)DIM >> 16)
              | (((uint32_t)tensor_rows & 0xFFFFu) << 16)); // tensor_dim1 lo
  g1[3] = (int)(((uint32_t)tensor_rows >> 16)
              | ((uint32_t)KCHUNK << 16));          // tile_dim0 = 128
  g1[4] = rows;                                     // tile_dim1, tile_dim2=0
  g1[5] = DIM;                                      // tensor_dim0_stride = 512
  g1[6] = 0;
  g1[7] = 0;
  i32x4 gz  = {0, 0, 0, 0};
  i32x8 gz8 = {0, 0, 0, 0, 0, 0, 0, 0};
  __builtin_amdgcn_tensor_load_to_lds(g0, g1, gz, gz, gz8, 0);
}

// ===========================================================================
// K1: gather (B,C,F,H,W) -> token rows, LayerNorm over C, cast to bf16.
// ===========================================================================
__global__ void ln_gather_kernel(const float* __restrict__ x,
                                 const float* __restrict__ ln_g,
                                 const float* __restrict__ ln_b,
                                 __bf16* __restrict__ xn) {
  const int tok  = blockIdx.x;
  const int site = tok >> 4, f = tok & 15;
  const int b = site >> 10, hw = site & 1023, hh = hw >> 5, ww = hw & 31;
  const long base = (long)b * DIM * CSTRIDE + (long)f * 1024 + hh * 32 + ww;

  const int t = threadIdx.x;                 // 128 threads, 4 channels each
  float v[4], s = 0.f, ss = 0.f;
#pragma unroll
  for (int i = 0; i < 4; ++i) {
    int c = t + i * 128;
    float val = x[base + (long)c * CSTRIDE];
    v[i] = val; s += val; ss += val * val;
  }
  __shared__ float rs[128], rq[128];
  rs[t] = s; rq[t] = ss;
  __syncthreads();
  for (int off = 64; off > 0; off >>= 1) {
    if (t < off) { rs[t] += rs[t + off]; rq[t] += rq[t + off]; }
    __syncthreads();
  }
  const float mu  = rs[0] * (1.f / DIM);
  const float var = rq[0] * (1.f / DIM) - mu * mu;
  const float inv = rsqrtf(var + 1e-5f);
#pragma unroll
  for (int i = 0; i < 4; ++i) {
    int c = t + i * 128;
    float y = (v[i] - mu) * inv * ln_g[c] + ln_b[c];
    xn[(long)tok * DIM + c] = (__bf16)y;
  }
}

// ===========================================================================
// K2: fp32 -> bf16 weight casts (W_qkv 1536x512, W_proj 512x512).
// ===========================================================================
__global__ void cvt_weights_kernel(const float* __restrict__ wqkv,
                                   const float* __restrict__ wproj,
                                   __bf16* __restrict__ wqkv_b,
                                   __bf16* __restrict__ wproj_b) {
  int i = blockIdx.x * 256 + threadIdx.x;
  const int NQ = 3 * DIM * DIM;
  if (i < NQ) wqkv_b[i] = (__bf16)wqkv[i];
  int j = i - NQ;
  if (j >= 0 && j < DIM * DIM) wproj_b[j] = (__bf16)wproj[j];
}

// ===========================================================================
// Shared GEMM core: rows [row0,row0+16) x cols [n0,n0+64), K = 512.
// B-panel (64 x 512 bf16) is streamed into LDS by the TDM in 4 double-
// buffered chunks of 128 K; A-fragments come from global (L2-resident),
// next A chunk software-prefetched. Per k-step: all 4 B-fragments are
// materialized first (one clause of 8 ds_load_b128, one dscnt wait), then
// 4 back-to-back v_wmma with reuse-A hints. 64 v_wmma per wave.
// ===========================================================================
__device__ inline void gemm_core_16x64(const __bf16* __restrict__ X,
                                       const __bf16* __restrict__ Wm,
                                       int wm_rows, int row0, int n0,
                                       __bf16* Blds0, __bf16* Blds1,
                                       v8f acc[4]) {
  const int nchunks = DIM / KCHUNK;           // 4
  __bf16* bufs[2] = {Blds0, Blds1};

  if (threadIdx.x < 32) {                     // wave 0 drives the TDM
    tdm_load_panel(Wm + (long)n0 * DIM,
                   (uint32_t)(uintptr_t)Blds0, BROWS, wm_rows);
    __builtin_amdgcn_s_wait_tensorcnt(0);
  }
  __syncthreads();

  for (int ch = 0; ch < nchunks; ++ch) {
    if (ch + 1 < nchunks && threadIdx.x < 32)
      tdm_load_panel(Wm + (long)n0 * DIM + (ch + 1) * KCHUNK,
                     (uint32_t)(uintptr_t)bufs[(ch + 1) & 1], BROWS, wm_rows);

    const __bf16* bl = bufs[ch & 1];
    const int kbase = ch * KCHUNK;
    // prefetch next A chunk for this wave's 16 rows
    if (ch + 1 < nchunks)
      __builtin_prefetch(X + (long)row0 * DIM + kbase + KCHUNK, 0, 1);
#pragma unroll
    for (int k0 = 0; k0 < KCHUNK; k0 += 32) {
      v16bf a  = load_a_frag(X, DIM, row0, kbase + k0);
      v16bf b0 = load_b_lds(bl,  0, k0);
      v16bf b1 = load_b_lds(bl, 16, k0);
      v16bf b2 = load_b_lds(bl, 32, k0);
      v16bf b3 = load_b_lds(bl, 48, k0);
      acc[0] = wmma_bf16   (a, b0, acc[0]);
      acc[1] = wmma_bf16_ra(a, b1, acc[1]);
      acc[2] = wmma_bf16_ra(a, b2, acc[2]);
      acc[3] = wmma_bf16_ra(a, b3, acc[3]);
    }
    if (ch + 1 < nchunks) {
      if (threadIdx.x < 32) __builtin_amdgcn_s_wait_tensorcnt(0);
      __syncthreads();
    }
  }
}

// ===========================================================================
// K3: QKV GEMM  qkv[tok, n] = sum_k xn[tok,k] * W_qkv[n,k]   (bf16 -> bf16)
// ===========================================================================
__global__ void __launch_bounds__(256)
gemm_qkv_kernel(const __bf16* __restrict__ X, const __bf16* __restrict__ Wm,
                __bf16* __restrict__ Y) {
  __shared__ __align__(16) __bf16 Blds[2][BROWS * BSTRIDE];   // 2 x 17 KiB
  const int wave = threadIdx.x >> 5;
  const int row0 = blockIdx.x * 128 + wave * 16;
  const int n0   = blockIdx.y * 64;
  v8f acc[4] = {v8f{}, v8f{}, v8f{}, v8f{}};
  gemm_core_16x64(X, Wm, 3 * DIM, row0, n0, Blds[0], Blds[1], acc);

  const int ln = threadIdx.x & 31, j = ln & 15, hi = ln >> 4;
#pragma unroll
  for (int nt = 0; nt < 4; ++nt)
#pragma unroll
    for (int e = 0; e < 8; ++e) {
      int m = e + hi * 8;
      Y[(long)(row0 + m) * (3 * DIM) + n0 + nt * 16 + j] = (__bf16)acc[nt][e];
    }
}

// ===========================================================================
// K4: attention. One wave per (site, head): block = 8 waves = 8 heads of one
// site. S = q k^T (2 bf16 WMMAs, K=64), softmax in accumulator layout
// (shfl_xor row-reductions within 16-lane halves), P and v^T staged in LDS,
// O = P v via 4 zero-padded (K=16->32) WMMAs sharing one A fragment.
// ===========================================================================
__global__ void __launch_bounds__(256)
attn_kernel(const __bf16* __restrict__ qkv, const float* __restrict__ rel_bias,
            __bf16* __restrict__ out) {
  __shared__ __bf16 vT[NHEADS][HDIM][NF];    // 16 KB : v transposed per head
  __shared__ __bf16 Pl[NHEADS][NF][NF];      //  4 KB : softmax probs per head

  const int site = blockIdx.x;
  const int head = threadIdx.x >> 5;
  const int ln   = threadIdx.x & 31;
  const int j    = ln & 15, hi = ln >> 4;

  const __bf16* qb = qkv + (long)site * NF * (3 * DIM) + head * HDIM;
  const __bf16* kb = qb + DIM;
  const __bf16* vb = qb + 2 * DIM;

  // stage v^T into LDS (1024 bf16 per wave)
  for (int idx = ln; idx < NF * HDIM; idx += 32) {
    int f = idx >> 6, d = idx & 63;
    vT[head][d][f] = vb[(long)f * (3 * DIM) + d];
  }

  // S = q k^T : rows = query frame, cols = key frame
  v8f s = {};
  s = wmma_bf16(load_a_frag(qb, 3 * DIM, 0, 0),
                load_b_frag(kb, 3 * DIM, 0, 0), s);
  s = wmma_bf16(load_a_frag(qb, 3 * DIM, 0, 32),
                load_b_frag(kb, 3 * DIM, 0, 32), s);

  // scale + rel_bias + row softmax (row i lives in one 16-lane half)
  const float scale = 0.125f;                 // 64^-0.5
#pragma unroll
  for (int e = 0; e < 8; ++e) {
    int i = e + hi * 8;
    float val = s[e] * scale + rel_bias[i * 64 + j];
    float mx = val;
#pragma unroll
    for (int msk = 1; msk <= 8; msk <<= 1)
      mx = fmaxf(mx, __shfl_xor(mx, msk, 32));
    float ex = __expf(val - mx);
    float sm = ex;
#pragma unroll
    for (int msk = 1; msk <= 8; msk <<= 1)
      sm += __shfl_xor(sm, msk, 32);
    float p = ex / sm;
    Pl[head][i][j] = (__bf16)p;
  }
  __syncthreads();

  // A-fragment of P, K padded 16 -> 32 with zeros
  FragU fa;
  fa.q[0] = hi ? make_float4(0.f, 0.f, 0.f, 0.f) : *(const float4*)&Pl[head][ln & 15][0];
  fa.q[1] = hi ? make_float4(0.f, 0.f, 0.f, 0.f) : *(const float4*)&Pl[head][ln & 15][8];

  // all 4 B-fragments first, then 4 WMMAs sharing A (reuse-A hints)
  FragU fb[4];
#pragma unroll
  for (int nt = 0; nt < 4; ++nt) {
    if (hi == 0) {                            // K=0..15 real, K=16..31 zero pad
      fb[nt].q[0] = *(const float4*)&vT[head][nt * 16 + j][0];
      fb[nt].q[1] = *(const float4*)&vT[head][nt * 16 + j][8];
    } else {
      fb[nt].q[0] = make_float4(0.f, 0.f, 0.f, 0.f);
      fb[nt].q[1] = make_float4(0.f, 0.f, 0.f, 0.f);
    }
  }
  v8f o0 = {}, o1 = {}, o2 = {}, o3 = {};
  o0 = wmma_bf16   (fa.v, fb[0].v, o0);
  o1 = wmma_bf16_ra(fa.v, fb[1].v, o1);
  o2 = wmma_bf16_ra(fa.v, fb[2].v, o2);
  o3 = wmma_bf16_ra(fa.v, fb[3].v, o3);

  __bf16* ob = out + (long)site * NF * DIM + head * HDIM;
  v8f os[4] = {o0, o1, o2, o3};
#pragma unroll
  for (int nt = 0; nt < 4; ++nt)
#pragma unroll
    for (int e = 0; e < 8; ++e) {
      int f = e + hi * 8;
      ob[(long)f * DIM + nt * 16 + j] = (__bf16)os[nt][e];
    }
}

// ===========================================================================
// K5: proj GEMM + bias + residual (re-read from x) + scatter to (B,C,F,H,W).
// ===========================================================================
__global__ void __launch_bounds__(256)
gemm_proj_kernel(const __bf16* __restrict__ A, const __bf16* __restrict__ Wm,
                 const float* __restrict__ bproj, const float* __restrict__ x,
                 float* __restrict__ out) {
  __shared__ __align__(16) __bf16 Blds[2][BROWS * BSTRIDE];
  const int wave = threadIdx.x >> 5;
  const int row0 = blockIdx.x * 128 + wave * 16;
  const int n0   = blockIdx.y * 64;
  v8f acc[4] = {v8f{}, v8f{}, v8f{}, v8f{}};
  gemm_core_16x64(A, Wm, DIM, row0, n0, Blds[0], Blds[1], acc);

  const int ln = threadIdx.x & 31, j = ln & 15, hi = ln >> 4;
#pragma unroll
  for (int nt = 0; nt < 4; ++nt) {
    const int c = n0 + nt * 16 + j;
    const float bc = bproj[c];
#pragma unroll
    for (int e = 0; e < 8; ++e) {
      int tok  = row0 + e + hi * 8;
      int site = tok >> 4, f = tok & 15;
      int b = site >> 10, hw = site & 1023, hh = hw >> 5, ww = hw & 31;
      long oi = ((((long)b * DIM + c) * NF + f) * 32 + hh) * 32 + ww;
      out[oi] = acc[nt][e] + bc + x[oi];     // residual read straight from x
    }
  }
}

// ===========================================================================
extern "C" void kernel_launch(void* const* d_in, const int* in_sizes, int n_in,
                              void* d_out, int out_size, void* d_ws, size_t ws_size,
                              hipStream_t stream) {
  const float* x        = (const float*)d_in[0];
  const float* W_qkv    = (const float*)d_in[1];
  const float* W_proj   = (const float*)d_in[2];
  const float* b_proj   = (const float*)d_in[3];
  const float* rel_bias = (const float*)d_in[4];
  const float* ln_g     = (const float*)d_in[5];
  const float* ln_b     = (const float*)d_in[6];
  float* out = (float*)d_out;

  // workspace layout (bytes)
  char* ws = (char*)d_ws;
  __bf16* xn      = (__bf16*)(ws);                               //  64 MiB
  __bf16* wqkv_b  = (__bf16*)(ws + (size_t)67108864);            // 1.5 MiB
  __bf16* wproj_b = (__bf16*)(ws + (size_t)68681728);            // 0.5 MiB
  __bf16* qkv     = (__bf16*)(ws + (size_t)69206016);            // 192 MiB
  __bf16* att     = (__bf16*)(ws + (size_t)270532608);           //  64 MiB

  ln_gather_kernel<<<NTOK, 128, 0, stream>>>(x, ln_g, ln_b, xn);

  const int NW = 3 * DIM * DIM + DIM * DIM;
  cvt_weights_kernel<<<(NW + 255) / 256, 256, 0, stream>>>(W_qkv, W_proj,
                                                           wqkv_b, wproj_b);

  gemm_qkv_kernel<<<dim3(NTOK / 128, (3 * DIM) / 64), 256, 0, stream>>>(
      xn, wqkv_b, qkv);

  attn_kernel<<<NSITES, 256, 0, stream>>>(qkv, rel_bias, att);

  gemm_proj_kernel<<<dim3(NTOK / 128, DIM / 64), 256, 0, stream>>>(
      att, wproj_b, b_proj, x, out);
}
